// GraphNetLayer_18056042512460
// MI455X (gfx1250) — matrix operand order repaired and checked by reading, not compiled, not run on hardware
//
#include <hip/hip_runtime.h>

#define H      256
#define H2     512
#define SEQ    4096
#define BSZ    32
#define MTOK   32    // tokens per block (2 WMMA M-tiles)
#define LDSTR  264   // padded bf16 row stride (528 B) to rotate LDS banks

typedef __bf16 bf16;
typedef __attribute__((ext_vector_type(16))) __bf16   v16bf;
typedef __attribute__((ext_vector_type(8)))  float    v8f;
typedef __attribute__((ext_vector_type(4)))  unsigned u32x4;

union FragU { u32x4 q[2]; v16bf v; };

__device__ __forceinline__ v16bf load_frag(const bf16* p) {
  FragU f;
  f.q[0] = *reinterpret_cast<const u32x4*>(p);
  f.q[1] = *reinterpret_cast<const u32x4*>(p + 16);
  return f.v;
}

// A-fragment (16x32 bf16, M x K) from row-major LDS matrix with padded stride.
// ISA layout: lanes 0-15 hold M=lane, K = kb+{0..7, 16..23}; lanes 16-31 K += 8.
__device__ __forceinline__ v16bf load_a(const bf16* mat, int row0, int kb, int lane) {
  const int m = lane & 15, half = lane >> 4;
  const bf16* p = mat + (row0 + m) * LDSTR + kb + 8 * half;
  return load_frag(p);
}

// B-fragment (32x16 bf16, K x N) from packed global weights:
// tile-major, 16 contiguous bf16 per lane (lane -> col lane&15, K-half lane>>4).
__device__ __forceinline__ v16bf load_b(const bf16* pw, int tile, int lane) {
  const bf16* p = pw + (((size_t)tile * 32 + lane) << 4);
  FragU f;
  f.q[0] = *reinterpret_cast<const u32x4*>(p);
  f.q[1] = *reinterpret_cast<const u32x4*>(p + 8);
  return f.v;
}

__device__ __forceinline__ v8f wmma_bf16(v16bf a, v16bf b, v8f c) {
  return __builtin_amdgcn_wmma_f32_16x16x32_bf16(false, a, false, b, (short)0, c,
                                                 false, false);
}

// Pack f32 W (K x N, row-major) into bf16 WMMA-B fragment order:
// dst[((kt*NT + nt)*32 + lane)*16 + e] = W[kt*32 + 16*(e>>3) + 8*(lane>>4) + (e&7)][nt*16 + (lane&15)]
__global__ void pack_weights(const float* __restrict__ W, bf16* __restrict__ dst,
                             int K, int N) {
  int idx = blockIdx.x * blockDim.x + threadIdx.x;
  if (idx >= K * N) return;
  int e    = idx & 15;
  int lane = (idx >> 4) & 31;
  int tile = idx >> 9;
  int NT   = N >> 4;
  int kt   = tile / NT, nt = tile - kt * NT;
  int k    = (kt << 5) + ((e >> 3) << 4) + ((lane >> 4) << 3) + (e & 7);
  int n    = (nt << 4) + (lane & 15);
  dst[idx] = (bf16)W[k * N + n];
}

__global__ __launch_bounds__(256)
void graphnet_kernel(const float* __restrict__ h,   const float* __restrict__ mask,
                     const float* __restrict__ eb1, const float* __restrict__ eb2,
                     const float* __restrict__ nb1, const float* __restrict__ nb2,
                     const bf16*  __restrict__ pe1, const bf16*  __restrict__ pe2,
                     const bf16*  __restrict__ pn1, const bf16*  __restrict__ pn2,
                     float* __restrict__ out) {
  __shared__ __align__(16) bf16 hRows[(MTOK + 2) * LDSTR];  // tokens t0-1 .. t0+32
  __shared__ __align__(16) bf16 hid  [MTOK * LDSTR];        // MLP hidden (bf16)
  __shared__ __align__(16) bf16 aggS [MTOK * LDSTR];        // agg as bf16 node input

  const int blk  = blockIdx.x;
  const int b    = blk >> 7;           // SEQ/32 = 128 tiles per sequence
  const int t0   = (blk & 127) << 5;
  const int tid  = threadIdx.x;
  const int lane = tid & 31;
  const int wave = tid >> 5;
  const int half = lane >> 4;
  const int ncol = lane & 15;
  const int nt0  = wave;               // wave's two N-tiles
  const int nt1  = wave + 8;

  // ---- Stage 0: stage h rows as bf16 (zeroed outside the sequence) ----
  for (int r = 0; r < MTOK + 2; ++r) {
    int t = t0 + r - 1;
    float v = 0.0f;
    if (t >= 0 && t < SEQ) v = h[((size_t)b * SEQ + t) * H + tid];
    hRows[r * LDSTR + tid] = (bf16)v;
  }
  __syncthreads();

  v8f agg[2][2] = {{v8f{}, v8f{}}, {v8f{}, v8f{}}};  // [msub][nsub]

  // ---- Edge MLP. pass 0: [h[t-1],h[t]] -> m_lr[t-1]; pass 1: [h[t+1],h[t]] -> m_rl[t]
  for (int pass = 0; pass < 2; ++pass) {
    // layer 1: (32x512) @ ew1 -> relu -> hid
    {
      v8f acc[2][2] = {{v8f{}, v8f{}}, {v8f{}, v8f{}}};
      #pragma unroll
      for (int kt = 0; kt < 16; ++kt) {
        const int kb   = kt << 5;
        const int row0 = (kb < H) ? (pass == 0 ? 0 : 2) : 1;
        const int col  = (kb < H) ? kb : (kb - H);
        v16bf a0 = load_a(hRows, row0,      col, lane);
        v16bf a1 = load_a(hRows, row0 + 16, col, lane);
        v16bf b0 = load_b(pe1, kt * 16 + nt0, lane);
        v16bf b1 = load_b(pe1, kt * 16 + nt1, lane);
        acc[0][0] = wmma_bf16(a0, b0, acc[0][0]);
        acc[1][0] = wmma_bf16(a1, b0, acc[1][0]);
        acc[0][1] = wmma_bf16(a0, b1, acc[0][1]);
        acc[1][1] = wmma_bf16(a1, b1, acc[1][1]);
      }
      #pragma unroll
      for (int nsub = 0; nsub < 2; ++nsub) {
        const int nt = wave + (nsub << 3);
        const float bias = eb1[nt * 16 + ncol];
        #pragma unroll
        for (int msub = 0; msub < 2; ++msub)
          #pragma unroll
          for (int r = 0; r < 8; ++r) {
            int m = (msub << 4) + r + (half << 3);
            hid[m * LDSTR + nt * 16 + ncol] = (bf16)fmaxf(acc[msub][nsub][r] + bias, 0.0f);
          }
      }
    }
    __syncthreads();
    // layer 2: hid @ ew2, boundary-masked, accumulate into agg registers
    {
      v8f acc[2][2] = {{v8f{}, v8f{}}, {v8f{}, v8f{}}};
      #pragma unroll
      for (int kt = 0; kt < 8; ++kt) {
        const int kb = kt << 5;
        v16bf a0 = load_a(hid, 0,  kb, lane);
        v16bf a1 = load_a(hid, 16, kb, lane);
        v16bf b0 = load_b(pe2, kt * 16 + nt0, lane);
        v16bf b1 = load_b(pe2, kt * 16 + nt1, lane);
        acc[0][0] = wmma_bf16(a0, b0, acc[0][0]);
        acc[1][0] = wmma_bf16(a1, b0, acc[1][0]);
        acc[0][1] = wmma_bf16(a0, b1, acc[0][1]);
        acc[1][1] = wmma_bf16(a1, b1, acc[1][1]);
      }
      #pragma unroll
      for (int nsub = 0; nsub < 2; ++nsub) {
        const int nt = wave + (nsub << 3);
        const float bias = eb2[nt * 16 + ncol];
        #pragma unroll
        for (int msub = 0; msub < 2; ++msub)
          #pragma unroll
          for (int r = 0; r < 8; ++r) {
            int t = t0 + (msub << 4) + r + (half << 3);
            bool valid = (pass == 0) ? (t > 0) : (t < SEQ - 1);
            agg[msub][nsub][r] += valid ? (acc[msub][nsub][r] + bias) : 0.0f;
          }
      }
    }
    __syncthreads();
  }

  // ---- agg -> bf16 LDS (second half of node-MLP input) ----
  #pragma unroll
  for (int nsub = 0; nsub < 2; ++nsub) {
    const int nt = wave + (nsub << 3);
    #pragma unroll
    for (int msub = 0; msub < 2; ++msub)
      #pragma unroll
      for (int r = 0; r < 8; ++r) {
        int m = (msub << 4) + r + (half << 3);
        aggS[m * LDSTR + nt * 16 + ncol] = (bf16)agg[msub][nsub][r];
      }
  }
  __syncthreads();

  // ---- Node MLP layer 1: [h, agg] (32x512) @ nw1 -> relu -> hid ----
  {
    v8f acc[2][2] = {{v8f{}, v8f{}}, {v8f{}, v8f{}}};
    #pragma unroll
    for (int kt = 0; kt < 16; ++kt) {
      const int kb = kt << 5;
      v16bf a0, a1;
      if (kb < H) { a0 = load_a(hRows, 1,  kb, lane);     a1 = load_a(hRows, 17, kb, lane); }
      else        { a0 = load_a(aggS,  0,  kb - H, lane); a1 = load_a(aggS,  16, kb - H, lane); }
      v16bf b0 = load_b(pn1, kt * 16 + nt0, lane);
      v16bf b1 = load_b(pn1, kt * 16 + nt1, lane);
      acc[0][0] = wmma_bf16(a0, b0, acc[0][0]);
      acc[1][0] = wmma_bf16(a1, b0, acc[1][0]);
      acc[0][1] = wmma_bf16(a0, b1, acc[0][1]);
      acc[1][1] = wmma_bf16(a1, b1, acc[1][1]);
    }
    #pragma unroll
    for (int nsub = 0; nsub < 2; ++nsub) {
      const int nt = wave + (nsub << 3);
      const float bias = nb1[nt * 16 + ncol];
      #pragma unroll
      for (int msub = 0; msub < 2; ++msub)
        #pragma unroll
        for (int r = 0; r < 8; ++r) {
          int m = (msub << 4) + r + (half << 3);
          hid[m * LDSTR + nt * 16 + ncol] = (bf16)fmaxf(acc[msub][nsub][r] + bias, 0.0f);
        }
    }
  }
  __syncthreads();

  // ---- Node MLP layer 2 + residual epilogue ----
  {
    v8f acc[2][2] = {{v8f{}, v8f{}}, {v8f{}, v8f{}}};
    #pragma unroll
    for (int kt = 0; kt < 8; ++kt) {
      const int kb = kt << 5;
      v16bf a0 = load_a(hid, 0,  kb, lane);
      v16bf a1 = load_a(hid, 16, kb, lane);
      v16bf b0 = load_b(pn2, kt * 16 + nt0, lane);
      v16bf b1 = load_b(pn2, kt * 16 + nt1, lane);
      acc[0][0] = wmma_bf16(a0, b0, acc[0][0]);
      acc[1][0] = wmma_bf16(a1, b0, acc[1][0]);
      acc[0][1] = wmma_bf16(a0, b1, acc[0][1]);
      acc[1][1] = wmma_bf16(a1, b1, acc[1][1]);
    }
    #pragma unroll
    for (int nsub = 0; nsub < 2; ++nsub) {
      const int nt = wave + (nsub << 3);
      const float bias = nb2[nt * 16 + ncol];
      const int n = nt * 16 + ncol;
      #pragma unroll
      for (int msub = 0; msub < 2; ++msub)
        #pragma unroll
        for (int r = 0; r < 8; ++r) {
          int t = t0 + (msub << 4) + r + (half << 3);
          size_t idx = ((size_t)b * SEQ + t) * H + n;
          float mk = mask[(size_t)b * SEQ + t];
          out[idx] = h[idx] + mk * (acc[msub][nsub][r] + bias);
        }
    }
  }
}

extern "C" void kernel_launch(void* const* d_in, const int* in_sizes, int n_in,
                              void* d_out, int out_size, void* d_ws, size_t ws_size,
                              hipStream_t stream) {
  const float* h    = (const float*)d_in[0];
  const float* mask = (const float*)d_in[1];
  const float* ew1  = (const float*)d_in[2];
  const float* eb1  = (const float*)d_in[3];
  const float* ew2  = (const float*)d_in[4];
  const float* eb2  = (const float*)d_in[5];
  const float* nw1  = (const float*)d_in[6];
  const float* nb1  = (const float*)d_in[7];
  const float* nw2  = (const float*)d_in[8];
  const float* nb2  = (const float*)d_in[9];
  float* out = (float*)d_out;

  // Workspace: bf16-packed weights (768 KB total), fully L2-resident.
  bf16* pe1 = reinterpret_cast<bf16*>(d_ws);
  bf16* pe2 = reinterpret_cast<bf16*>((char*)d_ws + (size_t)2 * H2 * H);
  bf16* pn1 = reinterpret_cast<bf16*>((char*)d_ws + (size_t)2 * H2 * H + (size_t)2 * H * H);
  bf16* pn2 = reinterpret_cast<bf16*>((char*)d_ws + (size_t)4 * H2 * H);

  pack_weights<<<(H2 * H + 255) / 256, 256, 0, stream>>>(ew1, pe1, H2, H);
  pack_weights<<<(H  * H + 255) / 256, 256, 0, stream>>>(ew2, pe2, H,  H);
  pack_weights<<<(H2 * H + 255) / 256, 256, 0, stream>>>(nw1, pn1, H2, H);
  pack_weights<<<(H  * H + 255) / 256, 256, 0, stream>>>(nw2, pn2, H,  H);

  graphnet_kernel<<<BSZ * (SEQ / MTOK), 256, 0, stream>>>(
      h, mask, eb1, eb2, nb1, nb2, pe1, pe2, pn1, pn2, out);
}
